// LDCA_84301618086462
// MI455X (gfx1250) — matrix operand order repaired and checked by reading, not compiled, not run on hardware
//
#include <hip/hip_runtime.h>
#include <math.h>

// ---------------- problem constants ----------------
#define B_    4
#define C_    256
#define H_    96
#define W_    96
#define HW_   9216          // H_*W_
#define NPIX_ 36864         // B_*HW_
#define HID_  64
#define G_    2
#define GC_   32
#define KS_   3
#define NTAP_ 9             // KS*KS
#define MLPH_ 1024
#define EPS_  1e-5f
#define SCALE_ 0.125f       // HID^-0.5
#define RF_   9.0f

typedef _Float16 half_t;
typedef _Float16 v16h  __attribute__((ext_vector_type(16)));
typedef float    v8f   __attribute__((ext_vector_type(8)));
typedef unsigned int uint4e __attribute__((ext_vector_type(4)));

union FragU { uint4e u[2]; v16h h; };

__device__ __forceinline__ float gelu_exact(float x) {
    return 0.5f * x * (1.0f + erff(x * 0.70710678118654752440f));
}

// ---------------- elementwise f32 -> f16 ----------------
__global__ void k_cvt_h(const float* __restrict__ in, half_t* __restrict__ out, int n) {
    int i = blockIdx.x * blockDim.x + threadIdx.x;
    if (i < n) out[i] = (half_t)in[i];
}

// ---------------- channel LayerNorm (NCHW), stats per pixel ----------------
__global__ void k_ln_chan_stats(const float* __restrict__ x, float* __restrict__ mean,
                                float* __restrict__ rstd) {
    int p = blockIdx.x * blockDim.x + threadIdx.x;
    if (p >= NPIX_) return;
    int b = p / HW_, hw = p % HW_;
    const float* base = x + (size_t)b * C_ * HW_ + hw;
    float s = 0.f, ss = 0.f;
    for (int c = 0; c < C_; ++c) { float v = base[(size_t)c * HW_]; s += v; ss += v * v; }
    float m = s * (1.0f / C_);
    float var = ss * (1.0f / C_) - m * m;
    mean[p] = m;
    rstd[p] = rsqrtf(var + EPS_);
}

// normalize + transpose NCHW f32 -> NHWC f16 via 16x16 LDS tile
__global__ void k_ln_chan_norm_t(const float* __restrict__ x, const float* __restrict__ mean,
                                 const float* __restrict__ rstd, const float* __restrict__ gamma,
                                 const float* __restrict__ beta, half_t* __restrict__ out) {
    __shared__ half_t tile[16][17];
    int b   = blockIdx.z;
    int c0  = blockIdx.y * 16;
    int hw0 = blockIdx.x * 16;
    int tx = threadIdx.x, ty = threadIdx.y;
    int c = c0 + ty, hw = hw0 + tx;
    int p = b * HW_ + hw;
    float v = x[((size_t)b * C_ + c) * HW_ + hw];
    v = (v - mean[p]) * rstd[p] * gamma[c] + beta[c];
    tile[ty][tx] = (half_t)v;
    __syncthreads();
    int pw = b * HW_ + hw0 + ty;
    out[(size_t)pw * C_ + c0 + tx] = tile[tx][ty];
}

// ---------------- row LayerNorm (last dim) -> f16, one wave per row ----------------
__global__ void k_ln_rows_h(const float* __restrict__ in, const float* __restrict__ gamma,
                            const float* __restrict__ beta, half_t* __restrict__ out,
                            int R, int D) {
    int wave = threadIdx.x >> 5, lane = threadIdx.x & 31;
    int row = blockIdx.x * (blockDim.x >> 5) + wave;
    if (row >= R) return;
    const float* rp = in + (size_t)row * D;
    float s = 0.f, ss = 0.f;
    for (int c = lane; c < D; c += 32) { float v = rp[c]; s += v; ss += v * v; }
    for (int o = 16; o > 0; o >>= 1) { s += __shfl_xor(s, o, 32); ss += __shfl_xor(ss, o, 32); }
    float m = s / (float)D;
    float rs = rsqrtf(ss / (float)D - m * m + EPS_);
    half_t* op = out + (size_t)row * D;
    for (int c = lane; c < D; c += 32)
        op[c] = (half_t)((rp[c] - m) * rs * gamma[c] + beta[c]);
}

// ---------------- WMMA GEMM: out[R,O] = X[R,K] * W[O,K]^T (+bias)(+res) ----------------
// res_mode: 0=none, 1=row-major [R,O], 2=NCHW [B,O,HW]
// out_mode: 0=row-major f32 [R,O], 1=NCHW f32 [B,O,HW]
__device__ __forceinline__ void gemm_store(const v8f& acc, int rowb, int col, int O,
                                           float* __restrict__ out,
                                           const float* __restrict__ bias,
                                           const float* __restrict__ res,
                                           int res_mode, int out_mode) {
    float bv = bias ? bias[col] : 0.0f;
#pragma unroll
    for (int r = 0; r < 8; ++r) {
        int row = rowb + r;
        float v = acc[r] + bv;
        if (res_mode == 1) {
            v += res[(size_t)row * O + col];
        } else if (res_mode == 2) {
            int bb = row / HW_, hw = row % HW_;
            v += res[((size_t)bb * O + col) * HW_ + hw];
        }
        if (out_mode == 0) {
            out[(size_t)row * O + col] = v;
        } else {
            int bb = row / HW_, hw = row % HW_;
            out[((size_t)bb * O + col) * HW_ + hw] = v;
        }
    }
}

__global__ void k_gemm_h(const half_t* __restrict__ X, const half_t* __restrict__ Wt,
                         float* __restrict__ out, const float* __restrict__ bias,
                         const float* __restrict__ res, int R, int O, int K,
                         int res_mode, int out_mode) {
    int lane = threadIdx.x & 31;
    int wave = threadIdx.x >> 5;
    int tiles_o = O >> 5;
    int t = blockIdx.x * (blockDim.x >> 5) + wave;
    int ntiles = (R >> 5) * tiles_o;
    if (t >= ntiles) return;                       // wave-uniform: EXEC all-1s for WMMA
    int rt = t / tiles_o, ct = t % tiles_o;
    int R0 = rt << 5, O0 = ct << 5;
    int hf = lane >> 4, lr = lane & 15;

    v8f acc00 = {0,0,0,0,0,0,0,0};
    v8f acc01 = {0,0,0,0,0,0,0,0};
    v8f acc10 = {0,0,0,0,0,0,0,0};
    v8f acc11 = {0,0,0,0,0,0,0,0};

    for (int kb = 0; kb < K; kb += 32) {
        FragU a0, a1, b0, b1;
        {   // A frag: lane (hf,lr) -> row R0[+16]+lr, K = kb + 8*hf + {0..7} and +16
            const half_t* p = X + (size_t)(R0 + lr) * K + kb + hf * 8;
            a0.u[0] = *(const uint4e*)(p);
            a0.u[1] = *(const uint4e*)(p + 16);
            p += (size_t)16 * K;
            a1.u[0] = *(const uint4e*)(p);
            a1.u[1] = *(const uint4e*)(p + 16);
        }
        {   // B frag: lane (hf,lr) -> col O0[+16]+lr, K = kb + 8*hf + {0..7} and +16
            const half_t* p = Wt + (size_t)(O0 + lr) * K + kb + hf * 8;
            b0.u[0] = *(const uint4e*)(p);
            b0.u[1] = *(const uint4e*)(p + 16);
            p += (size_t)16 * K;
            b1.u[0] = *(const uint4e*)(p);
            b1.u[1] = *(const uint4e*)(p + 16);
        }
        acc00 = __builtin_amdgcn_wmma_f32_16x16x32_f16(false, a0.h, false, b0.h, (short)0, acc00, false, false);
        acc01 = __builtin_amdgcn_wmma_f32_16x16x32_f16(false, a0.h, false, b1.h, (short)0, acc01, false, false);
        acc10 = __builtin_amdgcn_wmma_f32_16x16x32_f16(false, a1.h, false, b0.h, (short)0, acc10, false, false);
        acc11 = __builtin_amdgcn_wmma_f32_16x16x32_f16(false, a1.h, false, b1.h, (short)0, acc11, false, false);
    }
    gemm_store(acc00, R0 +      8 * hf, O0 +      lr, O, out, bias, res, res_mode, out_mode);
    gemm_store(acc01, R0 +      8 * hf, O0 + 16 + lr, O, out, bias, res, res_mode, out_mode);
    gemm_store(acc10, R0 + 16 + 8 * hf, O0 +      lr, O, out, bias, res, res_mode, out_mode);
    gemm_store(acc11, R0 + 16 + 8 * hf, O0 + 16 + lr, O, out, bias, res, res_mode, out_mode);
}

// ---------------- offset branch: depthwise3x3 + chanLN + GELU ----------------
__global__ void k_dw_ln_gelu(const float* __restrict__ qkv1, const float* __restrict__ dww,
                             const float* __restrict__ lng, const float* __restrict__ lnb,
                             float* __restrict__ tact) {
    int idx = blockIdx.x * blockDim.x + threadIdx.x;     // over B*G*HW
    if (idx >= B_ * G_ * HW_) return;
    int bg = idx / HW_, hw = idx % HW_;
    int b = bg >> 1, g = bg & 1;
    int i = hw / W_, j = hw % W_;
    float t[GC_];
    float s = 0.f, ss = 0.f;
    for (int cc = 0; cc < GC_; ++cc) {
        const float* plane = qkv1 + ((size_t)b * 192 + g * GC_ + cc) * HW_;
        float acc = 0.f;
#pragma unroll
        for (int ky = 0; ky < 3; ++ky) {
            int y = i + ky - 1;
            if (y < 0 || y >= H_) continue;
#pragma unroll
            for (int kx = 0; kx < 3; ++kx) {
                int x2 = j + kx - 1;
                if (x2 < 0 || x2 >= W_) continue;
                acc += dww[cc * 9 + ky * 3 + kx] * plane[y * W_ + x2];
            }
        }
        t[cc] = acc; s += acc; ss += acc * acc;
    }
    float m = s * (1.0f / GC_);
    float rs = rsqrtf(ss * (1.0f / GC_) - m * m + EPS_);
    for (int cc = 0; cc < GC_; ++cc) {
        float v = (t[cc] - m) * rs * lng[cc] + lnb[cc];
        tact[((size_t)bg * GC_ + cc) * HW_ + hw] = gelu_exact(v);
    }
}

// conv3x3 GC->18 + bias, then tanh*RF + base_offset
__global__ void k_off_conv(const float* __restrict__ tact, const float* __restrict__ w,
                           const float* __restrict__ bias, float* __restrict__ off) {
    int idx = blockIdx.x * blockDim.x + threadIdx.x;     // over B*G*18*HW
    if (idx >= B_ * G_ * 18 * HW_) return;
    int hw = idx % HW_;
    int rest = idx / HW_;
    int oc = rest % 18;
    int bg = rest / 18;
    int i = hw / W_, j = hw % W_;
    float acc = bias[oc];
    for (int cc = 0; cc < GC_; ++cc) {
        const float* plane = tact + ((size_t)bg * GC_ + cc) * HW_;
        const float* wk = w + ((size_t)oc * GC_ + cc) * 9;
#pragma unroll
        for (int ky = 0; ky < 3; ++ky) {
            int y = i + ky - 1;
            if (y < 0 || y >= H_) continue;
#pragma unroll
            for (int kx = 0; kx < 3; ++kx) {
                int x2 = j + kx - 1;
                if (x2 < 0 || x2 >= W_) continue;
                acc += wk[ky * 3 + kx] * plane[y * W_ + x2];
            }
        }
    }
    int tap = oc >> 1;
    float base = (oc & 1) ? (float)(tap % 3 - 1) : (float)(tap / 3 - 1);
    off[((size_t)bg * 18 + oc) * HW_ + hw] = tanhf(acc) * RF_ + base;
}

// ---------------- fused deformable sampling + 18-key attention ----------------
// one wave per pixel; lane owns channels d0=2*lane, d0+1 (group g = lane/16)
__global__ void k_attend(const float* __restrict__ qkv1, const float* __restrict__ kv2,
                         const float* __restrict__ off1, const float* __restrict__ off2,
                         const float* __restrict__ rpb, float* __restrict__ o1) {
    int wave = threadIdx.x >> 5, lane = threadIdx.x & 31;
    int p = blockIdx.x * (blockDim.x >> 5) + wave;
    if (p >= NPIX_) return;
    int b = p / HW_, hw = p % HW_;
    int i = hw / W_, j = hw % W_;
    int g = lane >> 4;
    int d0 = lane * 2;
    int cc0 = d0 & 31;
    int bg = b * 2 + g;

    float q0  = qkv1[((size_t)b * 192 + d0    ) * HW_ + hw] * SCALE_;
    float q1v = qkv1[((size_t)b * 192 + d0 + 1) * HW_ + hw] * SCALE_;

    float pl[18], vs0[18], vs1[18];

#pragma unroll
    for (int jt = 0; jt < 18; ++jt) {
        int br = jt / NTAP_, tap = jt % NTAP_;
        const float* offp = (br == 0 ? off1 : off2) + ((size_t)bg * 18 + tap * 2) * HW_ + hw;
        float ry = (float)i + offp[0];
        float rx = (float)j + offp[HW_];
        float y0f = floorf(ry), x0f = floorf(rx);
        float wy = ry - y0f, wx = rx - x0f;
        int y0 = (int)y0f, x0 = (int)x0f;
        int y1 = y0 + 1, x1 = x0 + 1;
        float w00 = (1.f - wy) * (1.f - wx), w01 = (1.f - wy) * wx;
        float w10 = wy * (1.f - wx),         w11 = wy * wx;
        float vy0 = (y0 >= 0 && y0 < H_) ? 1.f : 0.f;
        float vy1 = (y1 >= 0 && y1 < H_) ? 1.f : 0.f;
        float vx0 = (x0 >= 0 && x0 < W_) ? 1.f : 0.f;
        float vx1 = (x1 >= 0 && x1 < W_) ? 1.f : 0.f;
        w00 *= vy0 * vx0; w01 *= vy0 * vx1; w10 *= vy1 * vx0; w11 *= vy1 * vx1;
        int yc0 = y0 < 0 ? 0 : (y0 >= H_ ? H_ - 1 : y0);
        int yc1 = y1 < 0 ? 0 : (y1 >= H_ ? H_ - 1 : y1);
        int xc0 = x0 < 0 ? 0 : (x0 >= W_ ? W_ - 1 : x0);
        int xc1 = x1 < 0 ? 0 : (x1 >= W_ ? W_ - 1 : x1);
        int i00 = yc0 * W_ + xc0, i01 = yc0 * W_ + xc1;
        int i10 = yc1 * W_ + xc0, i11 = yc1 * W_ + xc1;

        const float *kp0, *vp0;
        if (br == 0) {
            kp0 = qkv1 + ((size_t)b * 192 +  64 + g * 32 + cc0) * HW_;
            vp0 = qkv1 + ((size_t)b * 192 + 128 + g * 32 + cc0) * HW_;
        } else {
            kp0 = kv2 + ((size_t)b * 128 +      g * 32 + cc0) * HW_;
            vp0 = kv2 + ((size_t)b * 128 + 64 + g * 32 + cc0) * HW_;
        }
        const float* kp1 = kp0 + HW_;
        const float* vp1 = vp0 + HW_;
        float k0 = w00 * kp0[i00] + w01 * kp0[i01] + w10 * kp0[i10] + w11 * kp0[i11];
        float k1 = w00 * kp1[i00] + w01 * kp1[i01] + w10 * kp1[i10] + w11 * kp1[i11];
        float v0 = w00 * vp0[i00] + w01 * vp0[i01] + w10 * vp0[i10] + w11 * vp0[i11];
        float v1 = w00 * vp1[i00] + w01 * vp1[i01] + w10 * vp1[i10] + w11 * vp1[i11];
        vs0[jt] = v0; vs1[jt] = v1;
        pl[jt] = q0  * (k0 + rpb[tap * 64 + d0]) +
                 q1v * (k1 + rpb[tap * 64 + d0 + 1]);
    }
#pragma unroll
    for (int jt = 0; jt < 18; ++jt) {
        float v = pl[jt];
        for (int o = 16; o > 0; o >>= 1) v += __shfl_xor(v, o, 32);
        pl[jt] = v;
    }
    float mx = pl[0];
#pragma unroll
    for (int jt = 1; jt < 18; ++jt) mx = fmaxf(mx, pl[jt]);
    float se = 0.f;
#pragma unroll
    for (int jt = 0; jt < 18; ++jt) { pl[jt] = expf(pl[jt] - mx); se += pl[jt]; }
    float inv = 1.0f / se;
    float oa = 0.f, ob = 0.f;
#pragma unroll
    for (int jt = 0; jt < 18; ++jt) { float a = pl[jt] * inv; oa += a * vs0[jt]; ob += a * vs1[jt]; }
    o1[(size_t)p * 64 + d0]     = oa;
    o1[(size_t)p * 64 + d0 + 1] = ob;
}

// ---------------- MLP depthwise3x3 + gelu(h+hd) -> f16 ----------------
__global__ void k_mlp_dw_gelu(const float* __restrict__ h, const float* __restrict__ dww,
                              const float* __restrict__ dwb, half_t* __restrict__ hact) {
    long long idx = (long long)blockIdx.x * blockDim.x + threadIdx.x;
    if (idx >= (long long)NPIX_ * MLPH_) return;
    int ch = (int)(idx % MLPH_);
    int p  = (int)(idx / MLPH_);
    int b = p / HW_, hw = p % HW_;
    int i = hw / W_, j = hw % W_;
    float acc = dwb[ch];
#pragma unroll
    for (int ky = 0; ky < 3; ++ky) {
        int y = i + ky - 1;
        if (y < 0 || y >= H_) continue;
#pragma unroll
        for (int kx = 0; kx < 3; ++kx) {
            int x2 = j + kx - 1;
            if (x2 < 0 || x2 >= W_) continue;
            acc += dww[ch * 9 + ky * 3 + kx] * h[((size_t)b * HW_ + y * W_ + x2) * MLPH_ + ch];
        }
    }
    float hv = h[(size_t)p * MLPH_ + ch];
    hact[(size_t)p * MLPH_ + ch] = (half_t)gelu_exact(hv + acc);
}

// ---------------- host launcher ----------------
extern "C" void kernel_launch(void* const* d_in, const int* in_sizes, int n_in,
                              void* d_out, int out_size, void* d_ws, size_t ws_size,
                              hipStream_t stream) {
    (void)in_sizes; (void)n_in; (void)out_size; (void)ws_size;
    const float* x        = (const float*)d_in[0];
    const float* y        = (const float*)d_in[1];
    const float* g11      = (const float*)d_in[2];
    const float* b11      = (const float*)d_in[3];
    const float* g21      = (const float*)d_in[4];
    const float* b21      = (const float*)d_in[5];
    const float* g12      = (const float*)d_in[6];
    const float* b12      = (const float*)d_in[7];
    const float* w_qkv1   = (const float*)d_in[8];
    const float* w_qkv2   = (const float*)d_in[9];
    const float* off1_dww = (const float*)d_in[10];
    const float* off1_lng = (const float*)d_in[11];
    const float* off1_lnb = (const float*)d_in[12];
    const float* off1_w   = (const float*)d_in[13];
    const float* off1_b   = (const float*)d_in[14];
    const float* off2_dww = (const float*)d_in[15];
    const float* off2_lng = (const float*)d_in[16];
    const float* off2_lnb = (const float*)d_in[17];
    const float* off2_w   = (const float*)d_in[18];
    const float* off2_b   = (const float*)d_in[19];
    const float* rpb      = (const float*)d_in[20];
    const float* proj_w   = (const float*)d_in[21];
    const float* proj_b   = (const float*)d_in[22];
    const float* fc1_w    = (const float*)d_in[23];
    const float* fc1_b    = (const float*)d_in[24];
    const float* dw_w     = (const float*)d_in[25];
    const float* dw_b     = (const float*)d_in[26];
    const float* fc2_w    = (const float*)d_in[27];
    const float* fc2_b    = (const float*)d_in[28];
    float* out = (float*)d_out;

    // workspace bump allocator (256B aligned)
    char* wsp = (char*)d_ws;
    auto alloc = [&](size_t bytes) -> void* {
        void* p = wsp;
        wsp += (bytes + 255) & ~(size_t)255;
        return p;
    };
    half_t* Xln    = (half_t*)alloc((size_t)NPIX_ * C_ * 2);
    half_t* Yln    = (half_t*)alloc((size_t)NPIX_ * C_ * 2);
    half_t* wq1h   = (half_t*)alloc((size_t)192 * 256 * 2);
    half_t* wq2h   = (half_t*)alloc((size_t)128 * 256 * 2);
    half_t* projwh = (half_t*)alloc((size_t)256 * 64 * 2);
    half_t* fc1wh  = (half_t*)alloc((size_t)1024 * 256 * 2);
    half_t* fc2wh  = (half_t*)alloc((size_t)256 * 1024 * 2);
    float* meanx = (float*)alloc((size_t)NPIX_ * 4);
    float* rstdx = (float*)alloc((size_t)NPIX_ * 4);
    float* meany = (float*)alloc((size_t)NPIX_ * 4);
    float* rstdy = (float*)alloc((size_t)NPIX_ * 4);
    float* qkv1  = (float*)alloc((size_t)B_ * 192 * HW_ * 4);
    float* kv2   = (float*)alloc((size_t)B_ * 128 * HW_ * 4);
    float* tact  = (float*)alloc((size_t)B_ * G_ * GC_ * HW_ * 4);
    float* off1  = (float*)alloc((size_t)B_ * G_ * 18 * HW_ * 4);
    float* off2  = (float*)alloc((size_t)B_ * G_ * 18 * HW_ * 4);
    float* o1    = (float*)alloc((size_t)NPIX_ * 64 * 4);
    half_t* o1h  = (half_t*)alloc((size_t)NPIX_ * 64 * 2);
    float* ores  = (float*)alloc((size_t)NPIX_ * C_ * 4);
    half_t* oresh = (half_t*)alloc((size_t)NPIX_ * C_ * 2);
    float* hbuf  = (float*)alloc((size_t)NPIX_ * MLPH_ * 4);
    half_t* hact = (half_t*)alloc((size_t)NPIX_ * MLPH_ * 2);

    auto cvt = [&](const float* src, half_t* dst, int n) {
        k_cvt_h<<<(n + 255) / 256, 256, 0, stream>>>(src, dst, n);
    };
    // weights -> f16
    cvt(w_qkv1, wq1h, 192 * 256);
    cvt(w_qkv2, wq2h, 128 * 256);
    cvt(proj_w, projwh, 256 * 64);
    cvt(fc1_w, fc1wh, 1024 * 256);
    cvt(fc2_w, fc2wh, 256 * 1024);

    // channel LN of x, y -> NHWC f16
    k_ln_chan_stats<<<NPIX_ / 256, 256, 0, stream>>>(x, meanx, rstdx);
    k_ln_chan_stats<<<NPIX_ / 256, 256, 0, stream>>>(y, meany, rstdy);
    {
        dim3 g(HW_ / 16, C_ / 16, B_), blk(16, 16);
        k_ln_chan_norm_t<<<g, blk, 0, stream>>>(x, meanx, rstdx, g11, b11, Xln);
        k_ln_chan_norm_t<<<g, blk, 0, stream>>>(y, meany, rstdy, g21, b21, Yln);
    }

    auto gemm = [&](const half_t* X, const half_t* Wt, float* o, const float* bias,
                    const float* res, int R, int O, int K, int res_mode, int out_mode) {
        int ntiles = (R / 32) * (O / 32);
        k_gemm_h<<<(ntiles + 3) / 4, 128, 0, stream>>>(X, Wt, o, bias, res, R, O, K,
                                                       res_mode, out_mode);
    };

    // qkv projections (output NCHW f32)
    gemm(Xln, wq1h, qkv1, nullptr, nullptr, NPIX_, 192, 256, 0, 1);
    gemm(Yln, wq2h, kv2,  nullptr, nullptr, NPIX_, 128, 256, 0, 1);

    // offset branches (both take q1 = qkv1[:, :64])
    {
        int n1 = B_ * G_ * HW_;
        int n2 = B_ * G_ * 18 * HW_;
        k_dw_ln_gelu<<<(n1 + 255) / 256, 256, 0, stream>>>(qkv1, off1_dww, off1_lng, off1_lnb, tact);
        k_off_conv<<<(n2 + 255) / 256, 256, 0, stream>>>(tact, off1_w, off1_b, off1);
        k_dw_ln_gelu<<<(n1 + 255) / 256, 256, 0, stream>>>(qkv1, off2_dww, off2_lng, off2_lnb, tact);
        k_off_conv<<<(n2 + 255) / 256, 256, 0, stream>>>(tact, off2_w, off2_b, off2);
    }

    // fused deformable sampling + attention (one wave per pixel)
    k_attend<<<NPIX_ / 4, 128, 0, stream>>>(qkv1, kv2, off1, off2, rpb, o1);

    // proj: ores = o1 @ proj_w^T + proj_b + x(NCHW residual)   [row-major out]
    cvt(o1, o1h, NPIX_ * 64);
    gemm(o1h, projwh, ores, proj_b, x, NPIX_, 256, 64, 2, 0);

    // MLP: h = ln_last(ores) @ fc1^T + fc1_b
    k_ln_rows_h<<<NPIX_ / 4, 128, 0, stream>>>(ores, g12, b12, oresh, NPIX_, C_);
    gemm(oresh, fc1wh, hbuf, fc1_b, nullptr, NPIX_, MLPH_, 256, 0, 0);

    // depthwise 3x3 + gelu(h + hd) -> f16
    {
        long long n = (long long)NPIX_ * MLPH_;
        k_mlp_dw_gelu<<<(unsigned)((n + 255) / 256), 256, 0, stream>>>(hbuf, dw_w, dw_b, hact);
    }

    // fc2 + fc2_b + ores residual, write final NCHW f32 straight to d_out
    gemm(hact, fc2wh, out, fc2_b, ores, NPIX_, 256, MLPH_, 1, 1);
}
